// HeterogeneousGraphClassifier_50070728737141
// MI455X (gfx1250) — matrix operand order repaired and checked by reading, not compiled, not run on hardware
//
#include <hip/hip_runtime.h>
#include <math.h>

typedef _Float16 v16h __attribute__((ext_vector_type(16)));
typedef _Float16 v8h  __attribute__((ext_vector_type(8)));
typedef float    v8f  __attribute__((ext_vector_type(8)));

#define DEVI static __device__ __forceinline__

DEVI unsigned fenc(float f){ unsigned u=__float_as_uint(f); return (u&0x80000000u)? ~u : (u|0x80000000u); }
DEVI float    fdec(unsigned u){ return __uint_as_float((u&0x80000000u)? (u&0x7fffffffu) : ~u); }

// ---------------- weight packing: f32 [K,N] -> f16 WMMA-B fragments ----------------
// fragment f = nTile*nK + kTile; per-fragment 32 lanes x 16 halves contiguous.
// lane n (0-15): col = nTile*16+n, K = kTile*32 + {0..7,16..23} ; lane 16+n: same col, K = kTile*32 + {8..15,24..31}
__global__ void pack_b_kernel(const float* __restrict__ W, _Float16* __restrict__ out, int K, int N){
  int nK = K >> 5;
  int f  = blockIdx.x;
  int kT = f % nK, nT = f / nK;
  int lane = threadIdx.x;
  int col = nT*16 + (lane & 15);
  int k0  = kT*32 + (lane >> 4)*8;
  _Float16* dst = out + ((size_t)f*32 + lane)*16;
  #pragma unroll
  for (int j=0;j<16;j++) dst[j] = (_Float16)W[(size_t)(k0+j+(j>=8?8:0))*N + col];
}

__global__ void cvt_f16_kernel(const float* __restrict__ in, _Float16* __restrict__ out, long n){
  long i = (long)blockIdx.x*blockDim.x + threadIdx.x;
  if (i<n) out[i] = (_Float16)in[i];
}

__global__ void gelu_f16_kernel(const float* __restrict__ in, _Float16* __restrict__ out, long n){
  long i = (long)blockIdx.x*blockDim.x + threadIdx.x;
  if (i<n){ float x=in[i]; out[i] = (_Float16)(0.5f*x*(1.f+erff(x*0.70710678118654752f))); }
}

// ---------------- WMMA GEMM: C[M,N] = A[M,K]f16 @ Bpacked + epilogue ----------------
// block = 64 threads (2 waves); each wave: one 32-row x 128-col C tile.
// A panel (2 fragments x NK k-tiles) preloaded once into registers; per N-tile we
// accumulate over K with only acc[2] live, then run the epilogue immediately.
// All epilogue configuration is compile-time: no branches in the unrolled tail.
// EPI: 0 = raw, 1 = +bias, 2 = relu(+bias), 3 = relu(ga*(C+bias) + (1-ga)*prev)
template<int NK, int EPI, bool WF, bool WH>
__launch_bounds__(64)
__global__ void gemm_wmma_kernel(const _Float16* __restrict__ A, int lda,
                                 const _Float16* __restrict__ Bp,
                                 const float* __restrict__ bias,
                                 const float* __restrict__ prev, int ldprev,
                                 const float* __restrict__ gate,
                                 float* __restrict__ outF, _Float16* __restrict__ outH,
                                 int ldo, int Mpairs){
  int lane = threadIdx.x & 31;
  int wave = threadIdx.x >> 5;
  int rp = blockIdx.x*2 + wave;                  // 32-row tile index
  if (rp >= Mpairs) return;                      // wave-uniform: EXEC stays all-ones for WMMA
  int s  = lane >> 4;                            // half-wave select
  int ml = lane & 15;

  // preload both 16-row A fragments for all K tiles (held in registers)
  v16h a[2][NK];
  const _Float16* abase = A + (size_t)(rp*32 + ml)*lda;
  #pragma unroll
  for (int h=0;h<2;h++){
    const _Float16* ab = abase + (size_t)(16*h)*lda;
    #pragma unroll
    for (int kt=0;kt<NK;kt++){
      const _Float16* ap = ab + kt*32 + s*8;
      v8h lo = *(const v8h*)(ap);
      v8h hi = *(const v8h*)(ap + 16);
      #pragma unroll
      for (int i=0;i<8;i++){ a[h][kt][i]=lo[i]; a[h][kt][i+8]=hi[i]; }
    }
  }

  float ga = 0.f;
  if (EPI==3) ga = 1.f/(1.f+__expf(-gate[0]));

  #pragma unroll
  for (int nt=0;nt<8;nt++){
    v8f acc[2];
    acc[0] = (v8f){0.f,0.f,0.f,0.f,0.f,0.f,0.f,0.f};
    acc[1] = (v8f){0.f,0.f,0.f,0.f,0.f,0.f,0.f,0.f};
    #pragma unroll
    for (int kt=0;kt<NK;kt++){
      int frag = (blockIdx.y*8 + nt)*NK + kt;
      v16h b = *(const v16h*)(Bp + ((size_t)frag*32 + lane)*16);
      acc[0] = __builtin_amdgcn_wmma_f32_16x16x32_f16(false, a[0][kt], false, b,
                                                      (short)0, acc[0], false, false);
      acc[1] = __builtin_amdgcn_wmma_f32_16x16x32_f16(false, a[1][kt], false, b,
                                                      (short)0, acc[1], false, false);
    }
    // epilogue for this 32x16 column tile (accumulators die here)
    int col = blockIdx.y*128 + nt*16 + ml;
    float bv = (EPI>=1) ? bias[col] : 0.f;
    #pragma unroll
    for (int h=0;h<2;h++){
      int rowBase = rp*32 + h*16 + s*8;
      #pragma unroll
      for (int r=0;r<8;r++){
        int row = rowBase + r;
        float v = acc[h][r] + bv;
        if (EPI==2) v = fmaxf(v, 0.f);
        if (EPI==3){
          float p = prev[(size_t)row*ldprev + col];
          v = fmaxf(ga*v + (1.f-ga)*p, 0.f);
        }
        size_t o = (size_t)row*ldo + col;
        if (WF) outF[o] = v;
        if (WH) outH[o] = (_Float16)v;
      }
    }
  }
}

// ---------------- edge phase ----------------
// wave per edge: logit = dot(q[di], ke[si]) * prel/sqrt(D); segment max via encoded atomicMax
__global__ void edge_logits_kernel(const _Float16* __restrict__ q, int ldq,
                                   const float* __restrict__ ke,
                                   const int* __restrict__ si, const int* __restrict__ di,
                                   int E, const float* __restrict__ prel,
                                   float* __restrict__ logits, unsigned* __restrict__ menc){
  int w = (int)(((long)blockIdx.x*blockDim.x + threadIdx.x) >> 5);
  int lane = threadIdx.x & 31;
  if (w >= E) return;
  int sx = si[w], dx = di[w];
  const _Float16* qp = q + (size_t)dx*ldq;
  const float*    kp = ke + (size_t)sx*128;
  float acc = 0.f;
  #pragma unroll
  for (int j=0;j<4;j++){ int d = lane + 32*j; acc += (float)qp[d]*kp[d]; }
  #pragma unroll
  for (int m=16;m;m>>=1) acc += __shfl_xor(acc, m, 32);
  if (lane==0){
    float lg = acc * (prel[0] * 0.088388347648318447f);  // 1/sqrt(128)
    logits[w] = lg;
    atomicMax(menc + dx, fenc(lg));
  }
}

__global__ void edge_exp_kernel(float* __restrict__ logits, const int* __restrict__ di, int E,
                                const unsigned* __restrict__ menc, float* __restrict__ ssum){
  int i = (int)((long)blockIdx.x*blockDim.x + threadIdx.x);
  if (i>=E) return;
  int d = di[i];
  float e = __expf(logits[i] - fdec(menc[d]));
  logits[i] = e;
  atomicAdd(ssum + d, e);
}

__global__ void edge_scatter_kernel(const float* __restrict__ expv,
                                    const int* __restrict__ si, const int* __restrict__ di, int E,
                                    const float* __restrict__ ssum, const float* __restrict__ ve,
                                    float* __restrict__ agg){
  int w = (int)(((long)blockIdx.x*blockDim.x + threadIdx.x) >> 5);
  int lane = threadIdx.x & 31;
  if (w >= E) return;
  int sx = si[w], dx = di[w];
  float alpha = expv[w] / (ssum[dx] + 1e-16f);
  const float* vp = ve + (size_t)sx*128;
  float* ap = agg + (size_t)dx*128;
  #pragma unroll
  for (int j=0;j<4;j++){ int d = lane + 32*j; atomicAdd(ap + d, alpha * vp[d]); }
}

// ---------------- pooling + head ----------------
__global__ void pool_kernel(const float* __restrict__ xp, const int* __restrict__ cb, int Nw,
                            float* __restrict__ pooled, float* __restrict__ cnt){
  int w = (int)(((long)blockIdx.x*blockDim.x + threadIdx.x) >> 5);
  int lane = threadIdx.x & 31;
  if (w >= Nw) return;
  int b = cb[w];
  const float* xr = xp + (size_t)w*128;
  #pragma unroll
  for (int j=0;j<4;j++){ int d = lane + 32*j; atomicAdd(pooled + b*128 + d, xr[d]); }
  if (lane==0) atomicAdd(cnt + b, 1.f);
}

__global__ void head_kernel(const float* __restrict__ pooled, const float* __restrict__ cnt,
                            const float* __restrict__ covs,
                            const float* __restrict__ W1, const float* __restrict__ b1,
                            const float* __restrict__ W2, const float* __restrict__ b2,
                            float* __restrict__ out, int B){
  __shared__ float h1[16*128];
  int tid = threadIdx.x;
  for (int idx=tid; idx<B*128; idx+=blockDim.x){
    int b = idx >> 7, j = idx & 127;
    float c = cnt[b]; c = (c < 1.f) ? 1.f : c;
    float a = b1[j];
    for (int i=0;i<128;i++) a += (pooled[b*128+i]/c) * W1[i*128+j];
    for (int i=0;i<8;i++)   a += covs[b*8+i] * W1[(128+i)*128+j];
    h1[idx] = fmaxf(a, 0.f);
  }
  __syncthreads();
  if (tid < B*2){
    int b = tid >> 1, c = tid & 1;
    float a = b2[c];
    for (int j=0;j<128;j++) a += h1[b*128+j] * W2[j*2+c];
    out[b*2+c] = a;
  }
}

// =====================================================================================
extern "C" void kernel_launch(void* const* d_in, const int* in_sizes, int n_in,
                              void* d_out, int out_size, void* d_ws, size_t ws_size,
                              hipStream_t stream){
  const float* x_gene    = (const float*)d_in[0];
  const float* x_protein = (const float*)d_in[1];
  const float* x_pathway = (const float*)d_in[2];
  const float* x_covs    = (const float*)d_in[3];
  const float* basis_W   = (const float*)d_in[4];
  const float* basis_b   = (const float*)d_in[5];
  const float* kqv_W     = (const float*)d_in[6];
  const float* kqv_b     = (const float*)d_in[7];
  const float* out_W     = (const float*)d_in[8];
  const float* out_b     = (const float*)d_in[9];
  const float* skip      = (const float*)d_in[10];
  const float* krel      = (const float*)d_in[11];
  const float* vrel      = (const float*)d_in[12];
  const float* prel      = (const float*)d_in[13];
  const float* lin1_W    = (const float*)d_in[14];
  const float* lin1_b    = (const float*)d_in[15];
  const float* lin2_W    = (const float*)d_in[16];
  const float* lin2_b    = (const float*)d_in[17];

  const int Ng = in_sizes[0]/64, Np = in_sizes[1]/64, Nw = in_sizes[2]/64;
  const int B  = in_sizes[3]/8;
  const int Nt[3] = {Ng, Np, Nw};
  const long off[3] = {0, (long)Ng, (long)Ng + Np};
  const long NTOT = (long)Ng + Np + Nw;

  const int Ecnt[4] = {in_sizes[18], in_sizes[20], in_sizes[22], in_sizes[24]};
  const int* esrc[4] = {(const int*)d_in[18], (const int*)d_in[20], (const int*)d_in[22], (const int*)d_in[24]};
  const int* edst[4] = {(const int*)d_in[19], (const int*)d_in[21], (const int*)d_in[23], (const int*)d_in[25]};
  const int* class_batch = (const int*)d_in[26];
  const int stype[4] = {0,1,1,1}, dtype[4] = {1,0,1,2};
  long eoff[4]; long ETOT = 0;
  for (int e=0;e<4;e++){ eoff[e]=ETOT; ETOT += Ecnt[e]; }

  // ---- workspace layout ----
  char* ws = (char*)d_ws;
  size_t cur = 0;
  auto alloc = [&](size_t bytes)->char*{ char* p = ws + cur; cur += (bytes + 255) & ~(size_t)255; return p; };
  _Float16* x16    = (_Float16*)alloc((size_t)NTOT*64*2);
  float*    xs32   = (float*)   alloc((size_t)NTOT*128*4);
  _Float16* xs16   = (_Float16*)alloc((size_t)NTOT*128*2);
  _Float16* kqv16  = (_Float16*)alloc((size_t)NTOT*384*2);
  float*    rel32  = (float*)   alloc((size_t)((Ng>Np?Ng:Np))*128*4);
  float*    agg32  = (float*)   alloc((size_t)NTOT*128*4);
  _Float16* agg16  = (_Float16*)alloc((size_t)NTOT*128*2);
  float*    logits = (float*)   alloc((size_t)ETOT*4);
  unsigned* menc   = (unsigned*)alloc((size_t)NTOT*4);
  float*    ssum   = (float*)   alloc((size_t)NTOT*4);
  float*    pooled = (float*)   alloc((size_t)B*128*4);
  float*    cntb   = (float*)   alloc((size_t)B*4);
  _Float16* pBasis = (_Float16*)alloc((size_t)3*64*128*2);
  _Float16* pKqv   = (_Float16*)alloc((size_t)6*128*384*2);
  _Float16* pOut   = (_Float16*)alloc((size_t)6*128*128*2);
  _Float16* pKrel  = (_Float16*)alloc((size_t)8*128*128*2);
  _Float16* pVrel  = (_Float16*)alloc((size_t)8*128*128*2);
  (void)ws_size; (void)n_in; (void)out_size;

  // ---- 1. pack all weights to WMMA fragment layout ----
  for (int t=0;t<3;t++)
    pack_b_kernel<<<(64/32)*(128/16), 32, 0, stream>>>(basis_W + (size_t)t*64*128, pBasis + (size_t)t*64*128, 64, 128);
  for (int i=0;i<6;i++)
    pack_b_kernel<<<(128/32)*(384/16), 32, 0, stream>>>(kqv_W + (size_t)i*128*384, pKqv + (size_t)i*128*384, 128, 384);
  for (int i=0;i<6;i++)
    pack_b_kernel<<<(128/32)*(128/16), 32, 0, stream>>>(out_W + (size_t)i*128*128, pOut + (size_t)i*128*128, 128, 128);
  for (int i=0;i<8;i++)
    pack_b_kernel<<<(128/32)*(128/16), 32, 0, stream>>>(krel + (size_t)i*128*128, pKrel + (size_t)i*128*128, 128, 128);
  for (int i=0;i<8;i++)
    pack_b_kernel<<<(128/32)*(128/16), 32, 0, stream>>>(vrel + (size_t)i*128*128, pVrel + (size_t)i*128*128, 128, 128);

  // ---- 2. convert raw inputs to f16 ----
  const float* xin[3] = {x_gene, x_protein, x_pathway};
  for (int t=0;t<3;t++){
    long n = (long)Nt[t]*64;
    cvt_f16_kernel<<<(unsigned)((n+255)/256), 256, 0, stream>>>(xin[t], x16 + off[t]*64, n);
  }

  // ---- 3. basis: xs = relu(x @ basis_W + b) ----  (NK=2, bias+relu, write f32+f16)
  for (int t=0;t<3;t++){
    dim3 g((unsigned)((Nt[t]/32+1)/2), 1, 1);
    gemm_wmma_kernel<2,2,true,true><<<g, 64, 0, stream>>>(
        x16 + off[t]*64, 64, pBasis + (size_t)t*64*128,
        basis_b + (size_t)t*128, nullptr, 0, nullptr,
        xs32 + off[t]*128, xs16 + off[t]*128, 128, Nt[t]/32);
  }

  // ---- 4. HGT layers ----
  for (int l=0;l<2;l++){
    hipMemsetAsync(menc, 0, (size_t)NTOT*4, stream);
    hipMemsetAsync(ssum, 0, (size_t)NTOT*4, stream);
    hipMemsetAsync(agg32, 0, (size_t)NTOT*128*4, stream);

    // kqv per type (NK=4, bias only, write f16)
    for (int t=0;t<3;t++){
      dim3 g((unsigned)((Nt[t]/32+1)/2), 3, 1);
      gemm_wmma_kernel<4,1,false,true><<<g, 64, 0, stream>>>(
          xs16 + off[t]*128, 128, pKqv + (size_t)(l*3+t)*128*384,
          kqv_b + (size_t)(l*3+t)*384, nullptr, 0, nullptr,
          nullptr, kqv16 + off[t]*384, 384, Nt[t]/32);
    }

    // per edge type: k-rel transform (NK=4, raw, write f32) then logits + segment max
    for (int e=0;e<4;e++){
      int st = stype[e], dt = dtype[e];
      dim3 g((unsigned)((Nt[st]/32+1)/2), 1, 1);
      gemm_wmma_kernel<4,0,true,false><<<g, 64, 0, stream>>>(
          kqv16 + off[st]*384 + 0, 384, pKrel + (size_t)(l*4+e)*128*128,
          nullptr, nullptr, 0, nullptr, rel32, nullptr, 128, Nt[st]/32);
      long waves = Ecnt[e];
      edge_logits_kernel<<<(unsigned)((waves*32+255)/256), 256, 0, stream>>>(
          kqv16 + off[dt]*384 + 128, 384, rel32, esrc[e], edst[e], Ecnt[e],
          prel + (l*4+e), logits + eoff[e], menc + off[dt]);
    }
    // exp + segment sum
    for (int e=0;e<4;e++){
      int dt = dtype[e];
      edge_exp_kernel<<<(unsigned)((Ecnt[e]+255)/256), 256, 0, stream>>>(
          logits + eoff[e], edst[e], Ecnt[e], menc + off[dt], ssum + off[dt]);
    }
    // per edge type: v-rel transform then alpha-weighted scatter
    for (int e=0;e<4;e++){
      int st = stype[e], dt = dtype[e];
      dim3 g((unsigned)((Nt[st]/32+1)/2), 1, 1);
      gemm_wmma_kernel<4,0,true,false><<<g, 64, 0, stream>>>(
          kqv16 + off[st]*384 + 256, 384, pVrel + (size_t)(l*4+e)*128*128,
          nullptr, nullptr, 0, nullptr, rel32, nullptr, 128, Nt[st]/32);
      long waves = Ecnt[e];
      edge_scatter_kernel<<<(unsigned)((waves*32+255)/256), 256, 0, stream>>>(
          logits + eoff[e], esrc[e], edst[e], Ecnt[e],
          ssum + off[dt], rel32, agg32 + off[dt]*128);
    }

    // gelu + f16 convert
    {
      long n = NTOT*128;
      gelu_f16_kernel<<<(unsigned)((n+255)/256), 256, 0, stream>>>(agg32, agg16, n);
    }
    // out proj with gated skip (NK=4, skip epilogue, write f32+f16; in-place per-element RMW)
    for (int t=0;t<3;t++){
      dim3 g((unsigned)((Nt[t]/32+1)/2), 1, 1);
      gemm_wmma_kernel<4,3,true,true><<<g, 64, 0, stream>>>(
          agg16 + off[t]*128, 128, pOut + (size_t)(l*3+t)*128*128,
          out_b + (size_t)(l*3+t)*128, xs32 + off[t]*128, 128, skip + (l*3+t),
          xs32 + off[t]*128, xs16 + off[t]*128, 128, Nt[t]/32);
    }
  }

  // ---- 5. mean pool over pathway nodes ----
  hipMemsetAsync(pooled, 0, (size_t)B*128*4, stream);
  hipMemsetAsync(cntb,   0, (size_t)B*4, stream);
  pool_kernel<<<(unsigned)(((long)Nw*32+255)/256), 256, 0, stream>>>(
      xs32 + off[2]*128, class_batch, Nw, pooled, cntb);

  // ---- 6. MLP head ----
  head_kernel<<<1, 256, 0, stream>>>(pooled, cntb, x_covs, lin1_W, lin1_b, lin2_W, lin2_b,
                                     (float*)d_out, B);
}
